// GNNEncoder_60378650247271
// MI455X (gfx1250) — matrix operand order repaired and checked by reading, not compiled, not run on hardware
//
#include <hip/hip_runtime.h>
#include <hip/hip_bf16.h>

typedef __attribute__((ext_vector_type(16))) _Float16 v16h;
typedef __attribute__((ext_vector_type(8)))  _Float16 v8h;
typedef __attribute__((ext_vector_type(8)))  float    v8f;

#define HD   128   // hidden dim H
#define HD2  256   // 2H
#define BM   64    // node rows per block in the WMMA MLP kernel
#define GCNT 512   // number of graphs

// ---------------------------------------------------------------- utilities
__global__ void zero_f32_kernel(float* __restrict__ p, int n) {
    int i = blockIdx.x * blockDim.x + threadIdx.x;
    if (i < n) p[i] = 0.0f;
}

// ------------------------------------------------ 180-row combo projection
// h0 = concat(node_emb[a], comp_emb[b], pin_emb[p]) @ W + b, but only
// 5*6*6 = 180 distinct input rows exist -> precompute them all.
__global__ void build_combo_kernel(const float* __restrict__ ne,
                                   const float* __restrict__ ce,
                                   const float* __restrict__ pe,
                                   const float* __restrict__ W,   // [384,128]
                                   const float* __restrict__ b,   // [128]
                                   float* __restrict__ combo) {   // [180,128]
    int c = blockIdx.x;            // 0..179
    int j = threadIdx.x;           // 0..127
    int a  = c / 36;
    int bb = (c / 6) % 6;
    int p  = c % 6;
    float acc = b[j];
    for (int k = 0; k < HD; ++k) acc = fmaf(ne[a * HD + k],  W[(k        ) * HD + j], acc);
    for (int k = 0; k < HD; ++k) acc = fmaf(ce[bb * HD + k], W[(HD   + k ) * HD + j], acc);
    for (int k = 0; k < HD; ++k) acc = fmaf(pe[p * HD + k],  W[(2*HD + k ) * HD + j], acc);
    combo[c * HD + j] = acc;
}

__global__ void gather_h0_kernel(const int* __restrict__ x,
                                 const float* __restrict__ combo,
                                 float* __restrict__ h, int N) {
    int n = blockIdx.x;
    int j = threadIdx.x;
    int a  = x[n * 3 + 0]; a  = a  < 0 ? 0 : a;
    int bb = x[n * 3 + 1]; bb = bb < 0 ? 0 : bb;
    int p  = x[n * 3 + 2]; p  = p  < 0 ? 0 : p;
    int c = a * 36 + bb * 6 + p;
    h[(size_t)n * HD + j] = combo[c * HD + j];
}

// --------------------------------------------- weight conversion (f32->f16)
// W1 [128,256] -> W1t [256][128] (transposed, row = output col, contig K)
// W2 [256,128] -> W2t [128][256]
__global__ void convert_w_kernel(const float* __restrict__ W1,
                                 const float* __restrict__ W2,
                                 _Float16* __restrict__ W1t,
                                 _Float16* __restrict__ W2t) {
    int i = blockIdx.x * blockDim.x + threadIdx.x;
    if (i < HD * HD2) {
        int k = i / HD2, n = i % HD2;
        W1t[n * HD + k] = (_Float16)W1[i];
    } else if (i < 2 * HD * HD2) {
        int j = i - HD * HD2;
        int k = j / HD, n = j % HD;
        W2t[n * HD2 + k] = (_Float16)W2[j];
    }
}

// ---------------------------------------------------- edge scatter (GIN agg)
// one wave per edge: 32 lanes x float4 = 128 features
__global__ void scatter_agg_kernel(const int* __restrict__ ei,
                                   const float* __restrict__ h,
                                   float* __restrict__ agg, int E) {
    int e = blockIdx.x * (blockDim.x >> 5) + (threadIdx.x >> 5);
    if (e >= E) return;
    int lane = threadIdx.x & 31;
    int src = ei[e];
    int dst = ei[E + e];
    const float4 v = ((const float4*)(h + (size_t)src * HD))[lane];
    float* ad = agg + (size_t)dst * HD + lane * 4;
    atomicAdd(ad + 0, v.x);
    atomicAdd(ad + 1, v.y);
    atomicAdd(ad + 2, v.z);
    atomicAdd(ad + 3, v.w);
}

// --------------------------------------------------- fused GIN MLP via WMMA
// z = h + agg (f16 in LDS); y1 = relu(z @ W1 + b1) kept in LDS (never hits
// HBM); zout = y1 @ W2 + b2 (f32, pre-batchnorm).
__global__ __launch_bounds__(256) void gin_mlp_kernel(
    const float* __restrict__ h, const float* __restrict__ agg,
    const _Float16* __restrict__ W1t, const float* __restrict__ b1,
    const _Float16* __restrict__ W2t, const float* __restrict__ b2,
    float* __restrict__ zout, int N) {
    __shared__ _Float16 sZ[BM * HD];    // 16 KB
    __shared__ _Float16 sY[BM * HD2];   // 32 KB
    int m0  = blockIdx.x * BM;
    int tid = threadIdx.x;

    // stage z = h + agg as f16
    for (int i = tid * 4; i < BM * HD; i += 256 * 4) {
        int r = i >> 7, c = i & (HD - 1);
        int n = m0 + r;
        float4 av, bv;
        if (n < N) {
            av = *(const float4*)(h   + (size_t)n * HD + c);
            bv = *(const float4*)(agg + (size_t)n * HD + c);
        } else {
            av = make_float4(0.f, 0.f, 0.f, 0.f);
            bv = av;
        }
        sZ[i + 0] = (_Float16)(av.x + bv.x);
        sZ[i + 1] = (_Float16)(av.y + bv.y);
        sZ[i + 2] = (_Float16)(av.z + bv.z);
        sZ[i + 3] = (_Float16)(av.w + bv.w);
    }
    __syncthreads();

    int wave = tid >> 5, lane = tid & 31;
    int lr = lane & 15, lh = lane >> 4;

    // GEMM1: [BM x 128] x [128 x 256] + b1, ReLU -> sY
    for (int t = wave; t < (BM / 16) * (HD2 / 16); t += 8) {
        int mt = t >> 4, nt = t & 15;
        float bias = b1[nt * 16 + lr];
        v8f acc;
#pragma unroll
        for (int j = 0; j < 8; ++j) acc[j] = bias;
#pragma unroll
        for (int k0 = 0; k0 < HD; k0 += 32) {
            // A fragment: lane row M = mt*16+lr; K = k0 + lh*16 .. +16 (contig)
            const _Float16* pa = sZ + (mt * 16 + lr) * HD;
            v16h A = *(const v16h*)(pa + k0 + lh * 16);
            // B fragment: lane col N = nt*16+lr, K = k0 + lh*16 .. +16 (contig)
            v16h B = *(const v16h*)(W1t + (nt * 16 + lr) * HD + k0 + lh * 16);
            acc = __builtin_amdgcn_wmma_f32_16x16x32_f16(
                false, A, false, B, (short)0, acc, false, false);
        }
#pragma unroll
        for (int j = 0; j < 8; ++j) {
            int row = mt * 16 + j + 8 * lh;     // C/D layout: VGPR j -> M=j (+8)
            sY[row * HD2 + nt * 16 + lr] = (_Float16)fmaxf(acc[j], 0.0f);
        }
    }
    __syncthreads();

    // GEMM2: [BM x 256] x [256 x 128] + b2 -> zout (f32)
    for (int t = wave; t < (BM / 16) * (HD / 16); t += 8) {
        int mt = t >> 3, nt = t & 7;
        float bias = b2[nt * 16 + lr];
        v8f acc;
#pragma unroll
        for (int j = 0; j < 8; ++j) acc[j] = bias;
#pragma unroll
        for (int k0 = 0; k0 < HD2; k0 += 32) {
            const _Float16* pa = sY + (mt * 16 + lr) * HD2;
            v16h A = *(const v16h*)(pa + k0 + lh * 16);
            v16h B = *(const v16h*)(W2t + (nt * 16 + lr) * HD2 + k0 + lh * 16);
            acc = __builtin_amdgcn_wmma_f32_16x16x32_f16(
                false, A, false, B, (short)0, acc, false, false);
        }
#pragma unroll
        for (int j = 0; j < 8; ++j) {
            int row = m0 + mt * 16 + j + 8 * lh;
            if (row < N) zout[(size_t)row * HD + nt * 16 + lr] = acc[j];
        }
    }
}

// ----------------------------------------------------------- batchnorm 1d
__global__ void bn_stats_kernel(const float* __restrict__ z,
                                float* __restrict__ stats, int N) {
    int c  = threadIdx.x;                  // 0..127
    int r0 = blockIdx.x * 256;
    int r1 = r0 + 256; if (r1 > N) r1 = N;
    float s = 0.f, s2 = 0.f;
    for (int r = r0; r < r1; ++r) {
        float v = z[(size_t)r * HD + c];
        s += v; s2 += v * v;
    }
    atomicAdd(&stats[c], s);
    atomicAdd(&stats[HD + c], s2);
}

__global__ void bn_apply_kernel(const float* __restrict__ z,
                                const float* __restrict__ stats,
                                const float* __restrict__ gamma,
                                const float* __restrict__ beta,
                                float* __restrict__ hbuf, int N, int residual) {
    int i = blockIdx.x * blockDim.x + threadIdx.x;
    if (i >= N * HD) return;
    int c = i & (HD - 1);
    float invN = 1.0f / (float)N;
    float mu  = stats[c] * invN;
    float var = stats[HD + c] * invN - mu * mu;   // population var == jnp.var
    float v = (z[i] - mu) * rsqrtf(var + 1e-5f) * gamma[c] + beta[c];
    v = fmaxf(v, 0.0f);
    if (residual) v += hbuf[i];
    hbuf[i] = v;
}

// ------------------------------------------------------------- readout
// out layout per graph g (384 cols): [0:128)=mean [128:256)=max [256:384)=sum
__global__ void readout_kernel(const float* __restrict__ h,
                               const int* __restrict__ batch,
                               float* __restrict__ out,
                               float* __restrict__ cnt, int N) {
    int n = blockIdx.x;
    int c = threadIdx.x;
    int g = batch[n];
    float v = h[(size_t)n * HD + c];
    atomicAdd(&out[(size_t)g * 384 + 256 + c], v);
    // h >= 0 after ReLU -> int-bit max is order-preserving; 0-init matches
    // the reference's isfinite->0 handling of empty segments
    atomicMax((int*)&out[(size_t)g * 384 + 128 + c], __float_as_int(v));
    if (c == 0) atomicAdd(&cnt[g], 1.0f);
}

__global__ void readout_final_kernel(float* __restrict__ out,
                                     const float* __restrict__ cnt) {
    int g = blockIdx.x;
    int c = threadIdx.x;
    float s = out[(size_t)g * 384 + 256 + c];
    out[(size_t)g * 384 + c] = s / fmaxf(cnt[g], 1.0f);
}

// ---------------------------------------------------------------- launcher
extern "C" void kernel_launch(void* const* d_in, const int* in_sizes, int n_in,
                              void* d_out, int out_size, void* d_ws, size_t ws_size,
                              hipStream_t stream) {
    const int*   x        = (const int*)d_in[0];
    const int*   ei       = (const int*)d_in[1];
    const int*   batch    = (const int*)d_in[2];
    const float* node_emb = (const float*)d_in[3];
    const float* comp_emb = (const float*)d_in[4];
    const float* pin_emb  = (const float*)d_in[5];
    const float* projW    = (const float*)d_in[6];
    const float* projb    = (const float*)d_in[7];

    const int N = in_sizes[0] / 3;
    const int E = in_sizes[1] / 2;

    // workspace carve-out
    char* ws = (char*)d_ws;
    size_t off = 0;
    auto take = [&](size_t bytes) -> char* {
        char* p = ws + off;
        off += (bytes + 255) & ~(size_t)255;
        return p;
    };
    float*    combo  = (float*)take(180 * HD * sizeof(float));
    float*    hbuf   = (float*)take((size_t)N * HD * sizeof(float));
    float*    aggbuf = (float*)take((size_t)N * HD * sizeof(float));
    float*    zbuf   = (float*)take((size_t)N * HD * sizeof(float));
    float*    stats  = (float*)take(2 * HD * sizeof(float));
    float*    cnt    = (float*)take(GCNT * sizeof(float));
    _Float16* W1t[3];
    _Float16* W2t[3];
    for (int l = 0; l < 3; ++l) {
        W1t[l] = (_Float16*)take((size_t)HD * HD2 * sizeof(_Float16));
        W2t[l] = (_Float16*)take((size_t)HD * HD2 * sizeof(_Float16));
    }

    // projection via 180-row combo table + gather
    build_combo_kernel<<<180, 128, 0, stream>>>(node_emb, comp_emb, pin_emb,
                                                projW, projb, combo);
    for (int l = 0; l < 3; ++l) {
        const float* W1 = (const float*)d_in[8 + 6 * l + 0];
        const float* W2 = (const float*)d_in[8 + 6 * l + 2];
        convert_w_kernel<<<(2 * HD * HD2 + 255) / 256, 256, 0, stream>>>(
            W1, W2, W1t[l], W2t[l]);
    }
    gather_h0_kernel<<<N, 128, 0, stream>>>(x, combo, hbuf, N);

    // 3 GIN layers
    for (int l = 0; l < 3; ++l) {
        const float* b1    = (const float*)d_in[8 + 6 * l + 1];
        const float* b2    = (const float*)d_in[8 + 6 * l + 3];
        const float* gamma = (const float*)d_in[8 + 6 * l + 4];
        const float* beta  = (const float*)d_in[8 + 6 * l + 5];

        zero_f32_kernel<<<(N * HD + 255) / 256, 256, 0, stream>>>(aggbuf, N * HD);
        scatter_agg_kernel<<<(E + 7) / 8, 256, 0, stream>>>(ei, hbuf, aggbuf, E);
        gin_mlp_kernel<<<(N + BM - 1) / BM, 256, 0, stream>>>(
            hbuf, aggbuf, W1t[l], b1, W2t[l], b2, zbuf, N);
        zero_f32_kernel<<<1, 256, 0, stream>>>(stats, 2 * HD);
        bn_stats_kernel<<<(N + 255) / 256, 128, 0, stream>>>(zbuf, stats, N);
        bn_apply_kernel<<<(N * HD + 255) / 256, 256, 0, stream>>>(
            zbuf, stats, gamma, beta, hbuf, N, l > 0);
    }

    // fused multi-pool readout
    zero_f32_kernel<<<(GCNT * 384 + 255) / 256, 256, 0, stream>>>((float*)d_out,
                                                                  GCNT * 384);
    zero_f32_kernel<<<(GCNT + 255) / 256, 256, 0, stream>>>(cnt, GCNT);
    readout_kernel<<<N, 128, 0, stream>>>(hbuf, batch, (float*)d_out, cnt, N);
    readout_final_kernel<<<GCNT, 128, 0, stream>>>((float*)d_out, cnt);
}
